// E81GateAsStateCell_3049426780854
// MI455X (gfx1250) — compile-verified, matrix operand order checked by reading, not run on hardware
//
#include <hip/hip_runtime.h>
#include <hip/hip_bf16.h>

#define T_STEPS 2048
#define BATCH   16
#define DDIM    1024
#define NDIM    64
#define PROJ_C  256                    // 4*NDIM
#define M_ROWS  (T_STEPS * BATCH)      // 32768

typedef __attribute__((ext_vector_type(16))) __bf16       v16bf;
typedef __attribute__((ext_vector_type(8)))  float        v8f;
typedef __attribute__((ext_vector_type(8)))  unsigned int v8u;

union Frag { v8u u; v16bf b; };

// Split an f32 pair into packed bf16 hi dword and bf16 lo (residual) dword.
__device__ __forceinline__ void split_pack2(float a, float b,
                                            unsigned int &hi, unsigned int &lo) {
  __bf16 ah = (__bf16)a;
  __bf16 bh = (__bf16)b;
  float  ra = a - (float)ah;
  float  rb = b - (float)bh;
  __bf16 al = (__bf16)ra;
  __bf16 bl = (__bf16)rb;
  hi = (unsigned int)__builtin_bit_cast(unsigned short, ah) |
       ((unsigned int)__builtin_bit_cast(unsigned short, bh) << 16);
  lo = (unsigned int)__builtin_bit_cast(unsigned short, al) |
       ((unsigned int)__builtin_bit_cast(unsigned short, bl) << 16);
}

// proj[M,256] = X[M,1024] @ W[256,1024]^T via split-bf16 WMMA (3 wmma / K-step).
__global__ __launch_bounds__(256) void e81_proj_wmma(
    const float* __restrict__ x, const float* __restrict__ W,
    float* __restrict__ proj)
{
  __shared__ unsigned int a_hi[128][17];   // stride 17 dwords: bank-conflict pad
  __shared__ unsigned int a_lo[128][17];
  __shared__ unsigned int b_hi[16][17];
  __shared__ unsigned int b_lo[16][17];

  const int tid  = threadIdx.x;
  const int wave = tid >> 5;
  const int lane = tid & 31;
  const int lm   = lane & 15;
  const int lh   = lane >> 4;

  const size_t row0 = (size_t)blockIdx.x * 128;
  const int    col0 = blockIdx.y * 16;

  v8f c = {};

  for (int kb = 0; kb < DDIM / 32; ++kb) {
    const int k0 = kb * 32;

    // Stage A: 128 rows x 32 cols f32, coalesced 128B per row-chunk.
    #pragma unroll
    for (int it = 0; it < 4; ++it) {
      const int slot = it * 256 + tid;        // 0..1023
      const int r    = slot >> 3;             // 0..127
      const int kq   = (slot & 7) * 4;        // 0..28
      const float4 f = *(const float4*)(x + (row0 + r) * DDIM + k0 + kq);
      split_pack2(f.x, f.y, a_hi[r][(kq >> 1) + 0], a_lo[r][(kq >> 1) + 0]);
      split_pack2(f.z, f.w, a_hi[r][(kq >> 1) + 1], a_lo[r][(kq >> 1) + 1]);
    }
    // Stage B: 16 rows (output channels) x 32 cols of W.
    if (tid < 128) {
      const int r  = tid >> 3;
      const int kq = (tid & 7) * 4;
      const float4 f = *(const float4*)(W + (size_t)(col0 + r) * DDIM + k0 + kq);
      split_pack2(f.x, f.y, b_hi[r][(kq >> 1) + 0], b_lo[r][(kq >> 1) + 0]);
      split_pack2(f.z, f.w, b_hi[r][(kq >> 1) + 1], b_lo[r][(kq >> 1) + 1]);
    }
    if (kb + 1 < DDIM / 32) {
      __builtin_prefetch(x + (row0 + (tid >> 1)) * DDIM + k0 + 32, 0, 1);
    }
    __syncthreads();

    // Fragment gather per documented 16-bit layouts (A 16x32, B mirrored).
    Frag ah, al, bh, bl;
    #pragma unroll
    for (int v = 0; v < 8; ++v) {
      const int di = (v & 3) + lh * 4 + (v >> 2) * 8;  // packed K-pair dword
      ah.u[v] = a_hi[wave * 16 + lm][di];
      al.u[v] = a_lo[wave * 16 + lm][di];
      bh.u[v] = b_hi[lm][di];
      bl.u[v] = b_lo[lm][di];
    }
    c = __builtin_amdgcn_wmma_f32_16x16x32_bf16(false, ah.b, false, bh.b,
                                                (short)0, c, false, false);
    c = __builtin_amdgcn_wmma_f32_16x16x32_bf16(false, ah.b, false, bl.b,
                                                (short)0, c, false, false);
    c = __builtin_amdgcn_wmma_f32_16x16x32_bf16(false, al.b, false, bh.b,
                                                (short)0, c, false, false);
    __syncthreads();
  }

  #pragma unroll
  for (int v = 0; v < 8; ++v) {
    const int m = v + lh * 8;                 // C/D layout: VGPRv = row m
    proj[(row0 + wave * 16 + m) * PROJ_C + col0 + lm] = c[v];
  }
}

__device__ __forceinline__ float sigmoidf(float v) {
  return 1.0f / (1.0f + __expf(-v));
}

// Sequential gated-state scan. One block per batch; S,G live in registers
// (16 elems/thread), kvqm staged through LDS, 4-lane shuffle reductions.
__global__ __launch_bounds__(256) void e81_scan(
    const float* __restrict__ proj,
    const float* __restrict__ S0, const float* __restrict__ G0,
    const float* __restrict__ b_s, const float* __restrict__ b_g,
    float* __restrict__ out, float* __restrict__ Sout, float* __restrict__ Gout)
{
  const int b   = blockIdx.x;
  const int tid = threadIdx.x;
  const int row = tid >> 2;            // 0..63
  const int jb  = (tid & 3) * 16;      // column base, 16 cols per thread

  float S[16], G[16];
  #pragma unroll
  for (int j = 0; j < 16; ++j) {
    S[j] = S0[(size_t)b * NDIM * NDIM + row * NDIM + jb + j];
    G[j] = G0[(size_t)b * NDIM * NDIM + row * NDIM + jb + j];
  }
  const float bsr = b_s[row];
  const float bgr = b_g[row];

  __shared__ float kv[PROJ_C];                  // [k|v|q|m]
  __shared__ float knorm[NDIM], mnorm[NDIM];
  __shared__ float sdelta[NDIM], red[NDIM];
  __shared__ float scal[2];

  for (int t = 0; t < T_STEPS; ++t) {
    kv[tid] = proj[(size_t)t * BATCH * PROJ_C + b * PROJ_C + tid];
    __syncthreads();

    // wave0: 1/(||k||+eps), wave1: 1/(||m||+eps)
    if (tid < 64) {
      const int w = tid >> 5, l = tid & 31;
      const float* src = (w == 0) ? kv : (kv + 192);
      float v0 = src[l], v1 = src[l + 32];
      float s = v0 * v0 + v1 * v1;
      #pragma unroll
      for (int o = 16; o > 0; o >>= 1) s += __shfl_xor(s, o, 32);
      if (l == 0) scal[w] = 1.0f / (sqrtf(s) + 1e-6f);
    }
    __syncthreads();
    if (tid < 64) {
      knorm[tid] = kv[tid]       * scal[0];
      mnorm[tid] = kv[192 + tid] * scal[1];
    }
    __syncthreads();

    // s_retrieved = S_old . k_norm
    float p = 0.f;
    #pragma unroll
    for (int j = 0; j < 16; ++j) p += S[j] * knorm[jb + j];
    p += __shfl_xor(p, 1, 32);
    p += __shfl_xor(p, 2, 32);
    if ((tid & 3) == 0) red[row] = p;
    __syncthreads();
    if (tid < 64) sdelta[tid] = kv[64 + tid] - red[tid];
    __syncthreads();

    // S = sigmoid(G_old + bs)*S + s_delta (x) k_norm
    const float sd = sdelta[row];
    #pragma unroll
    for (int j = 0; j < 16; ++j) {
      const float gs = sigmoidf(G[j] + bsr);
      S[j] = gs * S[j] + sd * knorm[jb + j];
    }

    // g_retrieved = G_old . m_norm
    p = 0.f;
    #pragma unroll
    for (int j = 0; j < 16; ++j) p += G[j] * mnorm[jb + j];
    p += __shfl_xor(p, 1, 32);
    p += __shfl_xor(p, 2, 32);
    if ((tid & 3) == 0) red[row] = p;
    __syncthreads();

    // G = sigmoid(S_new + bg)*G + g_delta (x) m_norm
    const float gd = sdelta[row] - red[row];
    #pragma unroll
    for (int j = 0; j < 16; ++j) {
      const float gg = sigmoidf(S[j] + bgr);
      G[j] = gg * G[j] + gd * mnorm[jb + j];
    }
    __syncthreads();   // all gd reads of red[] done before Sq reduce reuses it

    // Sq = S_new . q ; out = Sq^2 * sigmoid(Sq)
    p = 0.f;
    #pragma unroll
    for (int j = 0; j < 16; ++j) p += S[j] * kv[128 + jb + j];
    p += __shfl_xor(p, 1, 32);
    p += __shfl_xor(p, 2, 32);
    if ((tid & 3) == 0) red[row] = p;
    __syncthreads();
    if (tid < 64) {
      const float sq = red[tid];
      out[(size_t)t * BATCH * NDIM + b * NDIM + tid] = sq * sq * sigmoidf(sq);
    }
    __syncthreads();   // protect kv/red reuse in next step
  }

  #pragma unroll
  for (int j = 0; j < 16; ++j) {
    Sout[(size_t)b * NDIM * NDIM + row * NDIM + jb + j] = S[j];
    Gout[(size_t)b * NDIM * NDIM + row * NDIM + jb + j] = G[j];
  }
}

extern "C" void kernel_launch(void* const* d_in, const int* in_sizes, int n_in,
                              void* d_out, int out_size, void* d_ws, size_t ws_size,
                              hipStream_t stream) {
  (void)in_sizes; (void)n_in; (void)out_size; (void)ws_size;
  const float* x  = (const float*)d_in[0];   // (T,B,D)
  const float* S0 = (const float*)d_in[1];   // (B,n,n)
  const float* G0 = (const float*)d_in[2];   // (B,n,n)
  const float* W  = (const float*)d_in[3];   // (4n,D)
  const float* bs = (const float*)d_in[4];   // (n)
  const float* bg = (const float*)d_in[5];   // (n)
  float* out  = (float*)d_out;
  float* proj = (float*)d_ws;                // M_ROWS*256 f32 = 32 MB scratch

  dim3 gblk(M_ROWS / 128, PROJ_C / 16);      // (256, 16)
  e81_proj_wmma<<<gblk, 256, 0, stream>>>(x, W, proj);

  float* Sout = out + (size_t)T_STEPS * BATCH * NDIM;
  float* Gout = Sout + (size_t)BATCH * NDIM * NDIM;
  e81_scan<<<BATCH, 256, 0, stream>>>(proj, S0, G0, bs, bg, out, Sout, Gout);
}
